// AttnDecoder_1889785610886
// MI455X (gfx1250) — matrix operand order repaired and checked by reading, not compiled, not run on hardware
//
#include <hip/hip_runtime.h>
#include <hip/hip_bf16.h>
#include <math.h>

// ---------------- types / helpers ----------------
typedef __attribute__((ext_vector_type(16))) __bf16 v16bf;
typedef __attribute__((ext_vector_type(8)))  float  v8f;

union ABTile { uint4 u[2]; v16bf v; };   // 32 bytes: one wmma bf16 operand per lane

#define V_VOCAB 50257
#define HDIM    1024
#define SLEN    2048
#define KMERGE  2048
#define THREEH  3072

__device__ __forceinline__ unsigned short f2bf_rne(float f) {
  unsigned u = __float_as_uint(f);
  unsigned r = u + 0x7FFFu + ((u >> 16) & 1u);
  return (unsigned short)(r >> 16);
}
__device__ __forceinline__ float sigmoidf_(float x) { return 1.0f / (1.0f + __expf(-x)); }

// ---------------- one-time kernels ----------------
__global__ void k_init(const float* __restrict__ hidden, float* __restrict__ h, int* __restrict__ word) {
  int i = blockIdx.x * blockDim.x + threadIdx.x;
  if (i < HDIM) h[i] = hidden[i];
  if (i == 0) *word = 1;  // SOS
}

__global__ void k_f2bf4(const float4* __restrict__ src, ushort4* __restrict__ dst, int n4) {
  int stride = gridDim.x * blockDim.x;
  for (int i = blockIdx.x * blockDim.x + threadIdx.x; i < n4; i += stride) {
    float4 f = src[i];
    ushort4 o;
    o.x = f2bf_rne(f.x); o.y = f2bf_rne(f.y); o.z = f2bf_rne(f.z); o.w = f2bf_rne(f.w);
    dst[i] = o;
  }
}

// eproj[j][i] = sum_k e[j][k] * W_e[i][k] + b_a[i]   (2048 x 1024, K = 1024)  via WMMA bf16
__global__ void k_eproj_wmma(const unsigned short* __restrict__ eb,
                             const unsigned short* __restrict__ web,
                             const float* __restrict__ b_a,
                             float* __restrict__ eproj) {
  const int wave = (blockIdx.x * blockDim.x + threadIdx.x) >> 5;  // 0..8191
  const int lane = threadIdx.x & 31;
  const int jt = wave >> 6;   // 0..127  (2048/16)
  const int it = wave & 63;   // 0..63   (1024/16)
  const int j0 = jt * 16, i0 = it * 16;
  const int lr = lane & 15;
  const int hi = lane >> 4;   // 0 or 1
  const unsigned short* arow = eb  + (size_t)(j0 + lr) * HDIM;
  const unsigned short* brow = web + (size_t)(i0 + lr) * HDIM;
  v8f c = {0.f, 0.f, 0.f, 0.f, 0.f, 0.f, 0.f, 0.f};
  for (int k0 = 0; k0 < HDIM; k0 += 32) {
    ABTile a, b;
    const int abase = k0 + hi * 16;  // A: 16 contiguous K values
    a.u[0] = *(const uint4*)(arow + abase);
    a.u[1] = *(const uint4*)(arow + abase + 8);
    const int bbase = k0 + hi * 16;  // B: 16 contiguous K values
    b.u[0] = *(const uint4*)(brow + bbase);
    b.u[1] = *(const uint4*)(brow + bbase + 8);
    c = __builtin_amdgcn_wmma_f32_16x16x32_bf16(false, a.v, false, b.v, (short)0, c, false, false);
  }
  const float bav = b_a[i0 + lr];
  float* drow = eproj + (size_t)(j0 + hi * 8) * HDIM + i0 + lr;
#pragma unroll
  for (int m = 0; m < 8; ++m) drow[(size_t)m * HDIM] = c[m] + bav;
}

// ---------------- per-step kernels ----------------
// gi[r] = W_ih[r].x + b_ih[r] ;  gh[r] = W_hh[r].h + b_hh[r]   (wave per row, 6144 waves)
__global__ void k_gates(const float* __restrict__ W_ih, const float* __restrict__ W_hh,
                        const float* __restrict__ b_ih, const float* __restrict__ b_hh,
                        const float* __restrict__ emb_table, const int* __restrict__ word,
                        const float* __restrict__ h,
                        float* __restrict__ gi, float* __restrict__ gh) {
  const int gw   = (blockIdx.x * blockDim.x + threadIdx.x) >> 5;
  const int lane = threadIdx.x & 31;
  const bool isIH = gw < THREEH;
  const int row = isIH ? gw : gw - THREEH;
  const float* W   = isIH ? W_ih : W_hh;
  const float* vec = isIH ? (emb_table + (size_t)(*word) * HDIM) : h;
  const float4* wr = (const float4*)(W + (size_t)row * HDIM);
  const float4* vr = (const float4*)vec;
  float acc = 0.f;
#pragma unroll
  for (int kk = 0; kk < 8; ++kk) {
    float4 wv = wr[kk * 32 + lane];
    float4 xv = vr[kk * 32 + lane];
    acc += wv.x * xv.x + wv.y * xv.y + wv.z * xv.z + wv.w * xv.w;
  }
  for (int off = 16; off; off >>= 1) acc += __shfl_xor(acc, off, 32);
  if (lane == 0) {
    if (isIH) gi[row] = acc + b_ih[row];
    else      gh[row] = acc + b_hh[row];
  }
}

__global__ void k_gru(const float* __restrict__ gi, const float* __restrict__ gh,
                      float* __restrict__ h) {
  int i = blockIdx.x * blockDim.x + threadIdx.x;
  if (i >= HDIM) return;
  float r = sigmoidf_(gi[i]          + gh[i]);
  float z = sigmoidf_(gi[i + HDIM]   + gh[i + HDIM]);
  float n = tanhf   (gi[i + 2*HDIM] + r * gh[i + 2*HDIM]);
  h[i] = (1.f - z) * n + z * h[i];
}

// q[r] = W_q[r] . h   (wave per row, 1024 waves)
__global__ void k_qvec(const float* __restrict__ W_q, const float* __restrict__ h,
                       float* __restrict__ q) {
  const int row  = (blockIdx.x * blockDim.x + threadIdx.x) >> 5;
  const int lane = threadIdx.x & 31;
  const float4* wr = (const float4*)(W_q + (size_t)row * HDIM);
  const float4* vr = (const float4*)h;
  float acc = 0.f;
#pragma unroll
  for (int kk = 0; kk < 8; ++kk) {
    float4 wv = wr[kk * 32 + lane];
    float4 xv = vr[kk * 32 + lane];
    acc += wv.x * xv.x + wv.y * xv.y + wv.z * xv.z + wv.w * xv.w;
  }
  for (int off = 16; off; off >>= 1) acc += __shfl_xor(acc, off, 32);
  if (lane == 0) q[row] = acc;
}

// s[j] = sum_i v_a[i] * tanh(eproj[j][i] + q[i])   (wave per j, 2048 waves)
__global__ void k_scores(const float* __restrict__ eproj, const float* __restrict__ q,
                         const float* __restrict__ v_a, float* __restrict__ s) {
  const int j    = (blockIdx.x * blockDim.x + threadIdx.x) >> 5;
  const int lane = threadIdx.x & 31;
  const float4* er = (const float4*)(eproj + (size_t)j * HDIM);
  const float4* qr = (const float4*)q;
  const float4* vr = (const float4*)v_a;
  float acc = 0.f;
#pragma unroll
  for (int kk = 0; kk < 8; ++kk) {
    float4 e4 = er[kk * 32 + lane];
    float4 q4 = qr[kk * 32 + lane];
    float4 v4 = vr[kk * 32 + lane];
    acc += v4.x * tanhf(e4.x + q4.x) + v4.y * tanhf(e4.y + q4.y)
         + v4.z * tanhf(e4.z + q4.z) + v4.w * tanhf(e4.w + q4.w);
  }
  for (int off = 16; off; off >>= 1) acc += __shfl_xor(acc, off, 32);
  if (lane == 0) s[j] = acc;
}

__global__ void k_softmax(const float* __restrict__ s, float* __restrict__ w,
                          float* __restrict__ wout) {
  __shared__ float red[256];
  const int tid = threadIdx.x;
  float m = -INFINITY;
  for (int j = tid; j < SLEN; j += 256) m = fmaxf(m, s[j]);
  red[tid] = m; __syncthreads();
  for (int o = 128; o; o >>= 1) { if (tid < o) red[tid] = fmaxf(red[tid], red[tid + o]); __syncthreads(); }
  m = red[0]; __syncthreads();
  float sum = 0.f;
  for (int j = tid; j < SLEN; j += 256) sum += expf(s[j] - m);
  red[tid] = sum; __syncthreads();
  for (int o = 128; o; o >>= 1) { if (tid < o) red[tid] += red[tid + o]; __syncthreads(); }
  const float inv = 1.f / red[0];
  for (int j = tid; j < SLEN; j += 256) {
    float e = expf(s[j] - m) * inv;
    w[j] = e; wout[j] = e;
  }
}

// ctx[i] = sum_j w[j] * e[j][i]   (1024 threads total, coalesced over i)
__global__ void k_ctx(const float* __restrict__ w, const float* __restrict__ e,
                      float* __restrict__ ctx) {
  const int i = blockIdx.x * blockDim.x + threadIdx.x;
  float acc = 0.f;
  for (int j = 0; j < SLEN; ++j) acc += w[j] * e[(size_t)j * HDIM + i];
  ctx[i] = acc;
}

__global__ void k_merge(const float* __restrict__ h, const float* __restrict__ ctx,
                        unsigned short* __restrict__ mergeb,
                        unsigned long long* __restrict__ cell) {
  const int i = blockIdx.x * blockDim.x + threadIdx.x;
  if (i == 0) *cell = 0ull;  // reset argmax cell
  if (i < HDIM)             mergeb[i] = f2bf_rne(h[i]);
  else if (i < KMERGE)      mergeb[i] = f2bf_rne(ctx[i - HDIM]);
}

// logits GEMV via WMMA bf16: wave handles 16 vocab rows; argmax via packed u64 atomicMax
__global__ void k_logits_wmma(const unsigned short* __restrict__ woutb,
                              const unsigned short* __restrict__ mergeb,
                              const float* __restrict__ b_out,
                              unsigned long long* __restrict__ cell) {
  const int wave = (blockIdx.x * blockDim.x + threadIdx.x) >> 5;
  if (wave >= 3142) return;              // ceil(50257/16), wave-uniform exit
  const int lane = threadIdx.x & 31;
  const int lr = lane & 15;
  const int hi = lane >> 4;
  const int v0 = wave * 16;
  int row = v0 + lr; if (row > V_VOCAB - 1) row = V_VOCAB - 1;
  const unsigned short* brow = woutb + (size_t)row * KMERGE;
  v8f c = {0.f, 0.f, 0.f, 0.f, 0.f, 0.f, 0.f, 0.f};
  for (int k0 = 0; k0 < KMERGE; k0 += 32) {
    ABTile a, b;
    const int abase = k0 + hi * 16;      // broadcast merge vector into all 16 A rows
    a.u[0] = *(const uint4*)(mergeb + abase);
    a.u[1] = *(const uint4*)(mergeb + abase + 8);
    const int bbase = k0 + hi * 16;      // 16 contiguous bf16 of this vocab row
    __builtin_prefetch(brow + bbase + 32, 0, 1);
    b.u[0] = *(const uint4*)(brow + bbase);
    b.u[1] = *(const uint4*)(brow + bbase + 8);
    c = __builtin_amdgcn_wmma_f32_16x16x32_bf16(false, a.v, false, b.v, (short)0, c, false, false);
  }
  // all D rows identical (A rows identical): lane l holds logit for vocab v0 + (l&15)
  const int myv = v0 + lr;
  float logit = c[0];
  if (myv < V_VOCAB) logit += b_out[myv]; else logit = -INFINITY;
  unsigned ub  = __float_as_uint(logit);
  unsigned key = (ub & 0x80000000u) ? ~ub : (ub | 0x80000000u);  // order-preserving
  unsigned long long pk = ((unsigned long long)key << 32) | (unsigned)myv;
  for (int off = 16; off; off >>= 1) {
    unsigned long long other = __shfl_xor(pk, off, 32);
    if (other > pk) pk = other;
  }
  if (lane == 0) atomicMax(cell, pk);
}

__global__ void k_finalize(const unsigned long long* __restrict__ cell,
                           int* __restrict__ word, float* __restrict__ out_word) {
  if (threadIdx.x == 0) {
    int idx = (int)(unsigned)(*cell & 0xFFFFFFFFull);
    *word = idx;
    *out_word = (float)idx;
  }
}

// ---------------- launch ----------------
extern "C" void kernel_launch(void* const* d_in, const int* in_sizes, int n_in,
                              void* d_out, int out_size, void* d_ws, size_t ws_size,
                              hipStream_t stream) {
  // inputs in setup_inputs() order
  const float* hidden    = (const float*)d_in[0];   // (1,1,1024)
  const float* e         = (const float*)d_in[1];   // (1,2048,1024)
  const float* emb_table = (const float*)d_in[2];   // (50257,1024)
  const float* W_ih      = (const float*)d_in[3];   // (3072,1024)
  const float* W_hh      = (const float*)d_in[4];   // (3072,1024)
  const float* b_ih      = (const float*)d_in[5];
  const float* b_hh      = (const float*)d_in[6];
  const float* W_e       = (const float*)d_in[7];   // (1024,1024)
  const float* W_q       = (const float*)d_in[8];   // (1024,1024)
  const float* b_a       = (const float*)d_in[9];
  const float* v_a       = (const float*)d_in[10];
  const float* W_out     = (const float*)d_in[11];  // (50257,2048)
  const float* b_out     = (const float*)d_in[12];

  float* out = (float*)d_out;          // [0..24] words (as float), [25..] weights 25x2048

  // workspace layout (bytes)
  char* ws = (char*)d_ws;
  size_t off = 0;
  auto take = [&](size_t bytes) { size_t r = off; off = (off + bytes + 255) & ~(size_t)255; return r; };
  unsigned short* woutb  = (unsigned short*)(ws + take((size_t)V_VOCAB * KMERGE * 2)); // 206 MB
  unsigned short* eb     = (unsigned short*)(ws + take((size_t)SLEN * HDIM * 2));
  unsigned short* web    = (unsigned short*)(ws + take((size_t)HDIM * HDIM * 2));
  float*          eproj  = (float*)        (ws + take((size_t)SLEN * HDIM * 4));
  float*          h      = (float*)        (ws + take(HDIM * 4));
  float*          gi     = (float*)        (ws + take(THREEH * 4));
  float*          gh     = (float*)        (ws + take(THREEH * 4));
  float*          q      = (float*)        (ws + take(HDIM * 4));
  float*          sbuf   = (float*)        (ws + take(SLEN * 4));
  float*          watt   = (float*)        (ws + take(SLEN * 4));
  float*          ctx    = (float*)        (ws + take(HDIM * 4));
  unsigned short* mergeb = (unsigned short*)(ws + take(KMERGE * 2));
  int*                word = (int*)                (ws + take(64));
  unsigned long long* cell = (unsigned long long*)(ws + take(64));

  // ---- one-time prep ----
  k_init<<<4, 256, 0, stream>>>(hidden, h, word);
  k_f2bf4<<<2048, 256, 0, stream>>>((const float4*)e,   (ushort4*)eb,   (SLEN * HDIM) / 4);
  k_f2bf4<<<1024, 256, 0, stream>>>((const float4*)W_e, (ushort4*)web,  (HDIM * HDIM) / 4);
  k_f2bf4<<<4096, 256, 0, stream>>>((const float4*)W_out, (ushort4*)woutb, (V_VOCAB * KMERGE) / 4);
  k_eproj_wmma<<<1024, 256, 0, stream>>>(eb, web, b_a, eproj);  // 8192 waves

  // ---- 25 sequential decode steps ----
  for (int t = 0; t < 25; ++t) {
    k_gates<<<768, 256, 0, stream>>>(W_ih, W_hh, b_ih, b_hh, emb_table, word, h, gi, gh);
    k_gru<<<4, 256, 0, stream>>>(gi, gh, h);
    k_qvec<<<128, 256, 0, stream>>>(W_q, h, q);
    k_scores<<<256, 256, 0, stream>>>(eproj, q, v_a, sbuf);
    k_softmax<<<1, 256, 0, stream>>>(sbuf, watt, out + 25 + (size_t)t * SLEN);
    k_ctx<<<4, 256, 0, stream>>>(watt, e, ctx);
    k_merge<<<8, 256, 0, stream>>>(h, ctx, mergeb, cell);
    k_logits_wmma<<<393, 256, 0, stream>>>(woutb, mergeb, b_out, cell); // 3142 waves
    k_finalize<<<1, 32, 0, stream>>>(cell, word, out + t);
  }
}
